// MultiHeadAttention_17008070492283
// MI455X (gfx1250) — matrix operand order repaired and checked by reading, not compiled, not run on hardware
//
#include <hip/hip_runtime.h>

// ---------------------------------------------------------------------------
// MultiHeadAttention forward for MI455X (gfx1250, wave32, WMMA).
// All GEMMs use v_wmma_f32_16x16x32_bf16 (bf16 in, f32 accumulate).
// V is stored transposed [B,H,DEPTH,SEQ] so attn@V B-fragments are
// per-lane-contiguous (b128 loads) exactly like the Q.K^T phase.
// ---------------------------------------------------------------------------

typedef __attribute__((ext_vector_type(16))) __bf16 v16bf;
typedef __attribute__((ext_vector_type(8)))  float  v8f;

#define D_MODEL   1024
#define NUM_HEADS 16
#define DEPTH     64
#define BATCH     2
#define SEQ       2048

// A-fragment (16x32, 16-bit): lane L holds row M=L%16;
// element j maps to K = (j<8 ? j : j+8) + (L<16 ? 0 : 8)   [ISA 7.12.2]
__device__ __forceinline__ int frag_a_k(int lane, int j) {
  return (j < 8 ? j : j + 8) + ((lane < 16) ? 0 : 8);
}
// B-fragment (32x16, 16-bit): lane L holds col N=L%16;
// element j maps to K = (j<8 ? j : j+8) + (L<16 ? 0 : 8)
__device__ __forceinline__ int frag_b_k(int lane, int j) {
  return (j < 8 ? j : j + 8) + ((lane < 16) ? 0 : 8);
}

// ---------------------------------------------------------------------------
// Kernel 1: Yh = split_heads(X @ W + bias), bf16.
//   TRANSPOSE_OUT=false: [B, H, S, DEPTH]   (Q, K layout)
//   TRANSPOSE_OUT=true : [B, H, DEPTH, S]   (V layout, for contiguous B-frags)
// Block = 128 threads (4 waves); wave computes one 16x16 tile, block = 16x64.
// ---------------------------------------------------------------------------
template <bool TRANSPOSE_OUT>
__global__ __launch_bounds__(128) void proj_split_heads(
    const float* __restrict__ X, const float* __restrict__ W,
    const float* __restrict__ bias, __bf16* __restrict__ Yh) {
  const int lane = threadIdx.x & 31;
  const int wave = threadIdx.x >> 5;
  const int m0 = blockIdx.y * 16;               // output row tile
  const int n0 = blockIdx.x * 64 + wave * 16;   // output col tile (this wave)
  const int mr = m0 + (lane & 15);              // A row for this lane
  const int nc = n0 + (lane & 15);              // B/C col for this lane

  v8f c = {};
  for (int k0 = 0; k0 < D_MODEL; k0 += 32) {
    v16bf a, b;
#pragma unroll
    for (int j = 0; j < 16; ++j) {
      a[j] = (__bf16)X[(size_t)mr * D_MODEL + k0 + frag_a_k(lane, j)];
      b[j] = (__bf16)W[(size_t)(k0 + frag_b_k(lane, j)) * D_MODEL + nc];
    }
    c = __builtin_amdgcn_wmma_f32_16x16x32_bf16(false, a, false, b,
                                                (short)0, c, false, false);
  }

  const float bn = bias[nc];
  const int h = nc >> 6, dp = nc & 63;          // col -> (head, depth)
#pragma unroll
  for (int r = 0; r < 8; ++r) {
    const int row = m0 + r + ((lane < 16) ? 0 : 8);
    const int bb = row >> 11, s = row & 2047;   // row -> (batch, seq)
    const size_t hb = (size_t)(bb * NUM_HEADS + h);
    if (TRANSPOSE_OUT)
      Yh[(hb * DEPTH + dp) * SEQ + s] = (__bf16)(c[r] + bn);
    else
      Yh[(hb * SEQ + s) * DEPTH + dp] = (__bf16)(c[r] + bn);
  }
}

// ---------------------------------------------------------------------------
// Kernel 2: per (b, h, 16-row q-tile):
//   phase 1: scores[16][2048] = (Q Kt)/8 + mask*-1e9  (WMMA, 128KB LDS)
//   phase 2: softmax rows in LDS (float4), write attn f32 to global (b128)
//   phase 3: OutH tile = attn @ V; Sk split across 8 waves, LDS reduction
// Block = 256 threads (8 waves), dynamic LDS = 128KB scores + 4KB reduction.
// ---------------------------------------------------------------------------
__global__ __launch_bounds__(256) void attention_kernel(
    const __bf16* __restrict__ Qh, const __bf16* __restrict__ Kh,
    const __bf16* __restrict__ Vt, const float* __restrict__ mask,
    float* __restrict__ attn, __bf16* __restrict__ OutH) {
  extern __shared__ float sc[];                 // [16][SEQ] + [1024] reduction
  float* red = sc + 16 * SEQ;
  const int lane = threadIdx.x & 31;
  const int wave = threadIdx.x >> 5;
  const int qt = blockIdx.x & 127;              // q-tile (S/16 = 128)
  const int bh = blockIdx.x >> 7;               // fused (b,h)
  const int b = bh >> 4;
  const int h = bh & 15;
  const size_t head_base = (size_t)bh * SEQ * DEPTH;

  // ---- load Q fragment (16x64 -> two 16x32 k-steps), reused for all k-tiles
  v16bf aq[2];
  const int qrow = qt * 16 + (lane & 15);
#pragma unroll
  for (int st = 0; st < 2; ++st)
#pragma unroll
    for (int j = 0; j < 16; ++j)
      aq[st][j] =
          Qh[head_base + (size_t)qrow * DEPTH + st * 32 + frag_a_k(lane, j)];

  // ---- phase 1: scores = Q Kt, scaled + masked, into LDS
  for (int nt = wave; nt < SEQ / 16; nt += 8) {
    const int sk = nt * 16 + (lane & 15);       // B col = key position
    v8f c = {};
#pragma unroll
    for (int st = 0; st < 2; ++st) {
      v16bf bk;                                 // B = Kh^T : B[d][sk] = Kh[sk][d]
#pragma unroll
      for (int j = 0; j < 16; ++j)
        bk[j] = Kh[head_base + (size_t)sk * DEPTH + st * 32 + frag_b_k(lane, j)];
      c = __builtin_amdgcn_wmma_f32_16x16x32_bf16(false, aq[st], false, bk,
                                                  (short)0, c, false, false);
    }
    const float mval = mask[b * SEQ + sk] * -1e9f;
#pragma unroll
    for (int r = 0; r < 8; ++r) {
      const int row = r + ((lane < 16) ? 0 : 8);
      sc[row * SEQ + sk] = c[r] * 0.125f + mval;   // 1/sqrt(64) = 0.125
    }
  }
  __syncthreads();

  // ---- phase 2: softmax per row (wave handles 2 rows), float4 streams
#pragma unroll
  for (int rr = 0; rr < 2; ++rr) {
    const int row = wave * 2 + rr;
    float4* srow = (float4*)&sc[row * SEQ];     // SEQ/4 = 512 vectors
    float m = -3.4e38f;
    for (int c4 = lane; c4 < SEQ / 4; c4 += 32) {
      const float4 t = srow[c4];
      m = fmaxf(m, fmaxf(fmaxf(t.x, t.y), fmaxf(t.z, t.w)));
    }
#pragma unroll
    for (int off = 16; off > 0; off >>= 1) m = fmaxf(m, __shfl_xor(m, off, 32));
    float sum = 0.f;
    for (int c4 = lane; c4 < SEQ / 4; c4 += 32) {
      float4 t = srow[c4];
      t.x = __expf(t.x - m); t.y = __expf(t.y - m);
      t.z = __expf(t.z - m); t.w = __expf(t.w - m);
      srow[c4] = t;
      sum += (t.x + t.y) + (t.z + t.w);
    }
#pragma unroll
    for (int off = 16; off > 0; off >>= 1) sum += __shfl_xor(sum, off, 32);
    const float inv = 1.0f / sum;
    float4* arow =
        (float4*)(attn + ((size_t)bh * SEQ + qt * 16 + row) * SEQ);
    for (int c4 = lane; c4 < SEQ / 4; c4 += 32) {
      float4 t = srow[c4];
      t.x *= inv; t.y *= inv; t.z *= inv; t.w *= inv;
      srow[c4] = t;
      arow[c4] = t;                             // the 512MB b128 store stream
    }
  }
  __syncthreads();

  // ---- phase 3: OutH tile = attn[16 x 2048] @ V[2048 x 64]
  // 8 waves = 4 depth n-tiles x 2 Sk halves; V read from Vt (contiguous K).
  const int dp = (wave & 3) * 16 + (lane & 15); // output depth column
  const int ks0 = (wave >> 2) * 32;             // this wave's Sk half (32 steps)
  v8f o = {};
  for (int ks = ks0; ks < ks0 + 32; ++ks) {
    v16bf a, bv;
#pragma unroll
    for (int j = 0; j < 16; ++j) {
      a[j] = (__bf16)sc[(lane & 15) * SEQ + ks * 32 + frag_a_k(lane, j)];
      bv[j] = Vt[head_base + (size_t)dp * SEQ + ks * 32 + frag_b_k(lane, j)];
    }
    o = __builtin_amdgcn_wmma_f32_16x16x32_bf16(false, a, false, bv,
                                                (short)0, o, false, false);
  }
  __syncthreads();                              // all sc reads complete
  if (wave >= 4) {                              // upper half dumps partials
#pragma unroll
    for (int r = 0; r < 8; ++r)
      red[((wave & 3) * 8 + r) * 32 + lane] = o[r];
  }
  __syncthreads();
  if (wave < 4) {
#pragma unroll
    for (int r = 0; r < 8; ++r) {
      const float val = o[r] + red[(wave * 8 + r) * 32 + lane];
      const int s = qt * 16 + r + ((lane < 16) ? 0 : 8);
      OutH[((size_t)b * SEQ + s) * D_MODEL + h * DEPTH + dp] = (__bf16)val;
    }
  }
}

// ---------------------------------------------------------------------------
// Kernel 3: out = OutH(bf16) @ wo + bo, f32 result straight into d_out.
// ---------------------------------------------------------------------------
__global__ __launch_bounds__(128) void out_proj(
    const __bf16* __restrict__ A, const float* __restrict__ W,
    const float* __restrict__ bias, float* __restrict__ Out) {
  const int lane = threadIdx.x & 31;
  const int wave = threadIdx.x >> 5;
  const int m0 = blockIdx.y * 16;
  const int n0 = blockIdx.x * 64 + wave * 16;
  const int mr = m0 + (lane & 15);
  const int nc = n0 + (lane & 15);

  v8f c = {};
  for (int k0 = 0; k0 < D_MODEL; k0 += 32) {
    v16bf a, b;
#pragma unroll
    for (int j = 0; j < 16; ++j) {
      a[j] = A[(size_t)mr * D_MODEL + k0 + frag_a_k(lane, j)];
      b[j] = (__bf16)W[(size_t)(k0 + frag_b_k(lane, j)) * D_MODEL + nc];
    }
    c = __builtin_amdgcn_wmma_f32_16x16x32_bf16(false, a, false, b,
                                                (short)0, c, false, false);
  }
  const float bn = bias[nc];
#pragma unroll
  for (int r = 0; r < 8; ++r) {
    const int row = m0 + r + ((lane < 16) ? 0 : 8);
    Out[(size_t)row * D_MODEL + nc] = c[r] + bn;
  }
}

// ---------------------------------------------------------------------------
extern "C" void kernel_launch(void* const* d_in, const int* in_sizes, int n_in,
                              void* d_out, int out_size, void* d_ws,
                              size_t ws_size, hipStream_t stream) {
  (void)in_sizes; (void)n_in; (void)out_size; (void)ws_size;

  // setup_inputs() order: v, k, q, mask, wq, wk, wv, wo, bq, bk, bv, bo
  const float* v    = (const float*)d_in[0];
  const float* k    = (const float*)d_in[1];
  const float* q    = (const float*)d_in[2];
  const float* mask = (const float*)d_in[3];
  const float* wq   = (const float*)d_in[4];
  const float* wk   = (const float*)d_in[5];
  const float* wv   = (const float*)d_in[6];
  const float* wo   = (const float*)d_in[7];
  const float* bq   = (const float*)d_in[8];
  const float* bk   = (const float*)d_in[9];
  const float* bv   = (const float*)d_in[10];
  const float* bo   = (const float*)d_in[11];

  // workspace: 4 bf16 buffers of B*H*S*DEPTH (= B*S*D_MODEL) elements = 32MB
  const size_t head_elems = (size_t)BATCH * NUM_HEADS * SEQ * DEPTH;
  __bf16* Qh   = (__bf16*)d_ws;
  __bf16* Kh   = Qh + head_elems;
  __bf16* Vt   = Kh + head_elems;   // transposed [B,H,DEPTH,SEQ]
  __bf16* OutH = Vt + head_elems;

  // d_out layout: out [B,S,D_MODEL] f32, then attn [B,H,S,S] f32
  float* out  = (float*)d_out;
  float* attn = out + (size_t)BATCH * SEQ * D_MODEL;

  const dim3 gemm_grid(D_MODEL / 64, BATCH * SEQ / 16);
  proj_split_heads<false><<<gemm_grid, 128, 0, stream>>>(q, wq, bq, Qh);
  proj_split_heads<false><<<gemm_grid, 128, 0, stream>>>(k, wk, bk, Kh);
  proj_split_heads<true ><<<gemm_grid, 128, 0, stream>>>(v, wv, bv, Vt);

  const int attn_blocks = BATCH * NUM_HEADS * (SEQ / 16);          // 4096
  const size_t lds_bytes = (size_t)(16 * SEQ + 1024) * sizeof(float); // 132KB
  attention_kernel<<<attn_blocks, 256, lds_bytes, stream>>>(Qh, Kh, Vt, mask,
                                                            attn, OutH);

  out_proj<<<gemm_grid, 128, 0, stream>>>(OutH, wo, bo, out);
}
